// MemOSInspiredMemory_5351529251294
// MI455X (gfx1250) — compile-verified, matrix operand order checked by reading, not run on hardware
//
#include <hip/hip_runtime.h>
#include <math.h>

typedef __attribute__((ext_vector_type(16))) _Float16 v16h;
typedef __attribute__((ext_vector_type(8)))  _Float16 v8h;
typedef __attribute__((ext_vector_type(8)))  float    v8f;

#define SROWS 2048
#define DCOLS 1024
#define NMEM  32
#define KPATCH 32
#define KDIM  1027   // D + 3
#define KPAD  1056   // KDIM rounded up to multiple of 32
#define MPAD  48     // 33 rows (32 memories + query) padded to 3 WMMA tiles
#define NEG_BIG -3.402823466e38f

__device__ __forceinline__ float tclip(float x) {
    return truncf(fminf(fmaxf(x, -128.f), 127.f));
}

// ---------------- zero init ----------------
__global__ void k_zero(float* p, int n) {
    int i = blockIdx.x * blockDim.x + threadIdx.x;
    if (i < n) p[i] = 0.f;
}

// ------- fused per-row std (entropy) + per-column sum over S, one pass over memories -------
// grid: NMEM * (SROWS/32) blocks, 256 threads. Block handles 32 rows of one memory.
__global__ void k_rowstats(const float* __restrict__ mem,
                           float* __restrict__ entropy,
                           float* __restrict__ colsum) {
    __shared__ float sm1[8], sm2[8];
    int b = blockIdx.x;
    int n = b >> 6;              // / (SROWS/32)
    int srow0 = (b & 63) << 5;   // * 32
    int tid = threadIdx.x;
    int c0 = tid * 4;
    const float* basep = mem + (size_t)n * SROWS * DCOLS;
    float a0 = 0.f, a1 = 0.f, a2 = 0.f, a3 = 0.f;
    for (int r = 0; r < 32; ++r) {
        int s = srow0 + r;
        float4 v = *reinterpret_cast<const float4*>(basep + (size_t)s * DCOLS + c0);
        float s1 = v.x + v.y + v.z + v.w;
        float s2 = v.x * v.x + v.y * v.y + v.z * v.z + v.w * v.w;
        a0 += v.x; a1 += v.y; a2 += v.z; a3 += v.w;
        for (int o = 16; o >= 1; o >>= 1) {
            s1 += __shfl_xor(s1, o, 32);
            s2 += __shfl_xor(s2, o, 32);
        }
        int wid = tid >> 5, lane = tid & 31;
        if (lane == 0) { sm1[wid] = s1; sm2[wid] = s2; }
        __syncthreads();
        if (tid == 0) {
            float t1 = 0.f, t2 = 0.f;
            for (int w = 0; w < 8; ++w) { t1 += sm1[w]; t2 += sm2[w]; }
            float mean = t1 * (1.f / (float)DCOLS);
            float var = (t2 - t1 * mean) * (1.f / (float)(DCOLS - 1));   // ddof=1
            entropy[(size_t)n * SROWS + s] = sqrtf(fmaxf(var, 0.f));
        }
        __syncthreads();
    }
    atomicAdd(&colsum[n * DCOLS + c0 + 0], a0);
    atomicAdd(&colsum[n * DCOLS + c0 + 1], a1);
    atomicAdd(&colsum[n * DCOLS + c0 + 2], a2);
    atomicAdd(&colsum[n * DCOLS + c0 + 3], a3);
}

// ------- query column sums: 4 col-tiles x 16 row-chunks, atomics into zeroed qcolsum -------
__global__ void k_qcolsum(const float* __restrict__ q, float* __restrict__ qcolsum) {
    int cb = blockIdx.x & 3;
    int rc = blockIdx.x >> 2;
    int d = cb * 256 + threadIdx.x;
    int s0 = rc * 128;
    float a = 0.f;
    for (int s = s0; s < s0 + 128; ++s) a += q[(size_t)s * DCOLS + d];
    atomicAdd(&qcolsum[d], a);
}

// ------- iterative top-k: one block per slice, writes 0/1 selection mask -------
__global__ void k_topk_rows(const float* __restrict__ vals, float* __restrict__ selmask,
                            int len, int k) {
    __shared__ float sv[256];
    __shared__ int   si[256];
    int n = blockIdx.x;
    const float* v = vals + (size_t)n * len;
    float* mask = selmask + (size_t)n * len;
    int tid = threadIdx.x;
    for (int s = tid; s < len; s += blockDim.x) mask[s] = 0.f;
    __syncthreads();
    for (int it = 0; it < k; ++it) {
        float best = NEG_BIG; int besti = len;
        for (int s = tid; s < len; s += blockDim.x) {
            if (mask[s] == 0.f) {
                float x = v[s];
                if (x > best) { best = x; besti = s; }   // strict > keeps lowest index on ties
            }
        }
        sv[tid] = best; si[tid] = besti;
        __syncthreads();
        for (int off = 128; off >= 1; off >>= 1) {
            if (tid < off) {
                float ov = sv[tid + off]; int oi = si[tid + off];
                if (ov > sv[tid] || (ov == sv[tid] && oi < si[tid])) { sv[tid] = ov; si[tid] = oi; }
            }
            __syncthreads();
        }
        if (tid == 0 && si[0] < len) mask[si[0]] = 1.f;
        __syncthreads();
    }
}

// ------- aggregate = (query + sum_n sel*trunc(clip(mem_row))) / 33, + agg row entropy -------
__global__ void k_aggregate(const float* __restrict__ q, const float* __restrict__ mem,
                            const float* __restrict__ selmask,
                            float* __restrict__ agg, float* __restrict__ aggent) {
    __shared__ float sm1[8], sm2[8];
    int s = blockIdx.x;
    int tid = threadIdx.x;
    int c0 = tid * 4;
    float4 acc = *reinterpret_cast<const float4*>(q + (size_t)s * DCOLS + c0);
    for (int n = 0; n < NMEM; ++n) {
        if (selmask[(size_t)n * SROWS + s] != 0.f) {
            float4 v = *reinterpret_cast<const float4*>(mem + ((size_t)n * SROWS + s) * DCOLS + c0);
            acc.x += tclip(v.x); acc.y += tclip(v.y);
            acc.z += tclip(v.z); acc.w += tclip(v.w);
        }
    }
    const float inv = 1.f / 33.f;
    acc.x *= inv; acc.y *= inv; acc.z *= inv; acc.w *= inv;
    *reinterpret_cast<float4*>(agg + (size_t)s * DCOLS + c0) = acc;
    float s1 = acc.x + acc.y + acc.z + acc.w;
    float s2 = acc.x * acc.x + acc.y * acc.y + acc.z * acc.z + acc.w * acc.w;
    for (int o = 16; o >= 1; o >>= 1) {
        s1 += __shfl_xor(s1, o, 32);
        s2 += __shfl_xor(s2, o, 32);
    }
    int wid = tid >> 5, lane = tid & 31;
    if (lane == 0) { sm1[wid] = s1; sm2[wid] = s2; }
    __syncthreads();
    if (tid == 0) {
        float t1 = 0.f, t2 = 0.f;
        for (int w = 0; w < 8; ++w) { t1 += sm1[w]; t2 += sm2[w]; }
        float mean = t1 * (1.f / (float)DCOLS);
        float var = (t2 - t1 * mean) * (1.f / (float)(DCOLS - 1));
        aggent[s] = sqrtf(fmaxf(var, 0.f));
    }
}

// ------- recon output: selected rows -> trunc(clip(agg)), others 0 -------
__global__ void k_recon(const float* __restrict__ agg, const float* __restrict__ outsel,
                        float* __restrict__ out) {
    int s = blockIdx.x;
    int c0 = threadIdx.x * 4;
    float4 r = make_float4(0.f, 0.f, 0.f, 0.f);
    if (outsel[s] != 0.f) {
        float4 v = *reinterpret_cast<const float4*>(agg + (size_t)s * DCOLS + c0);
        r.x = tclip(v.x); r.y = tclip(v.y); r.z = tclip(v.z); r.w = tclip(v.w);
    }
    *reinterpret_cast<float4*>(out + (size_t)s * DCOLS + c0) = r;
}

// ------- operand-panel prep for WMMA (f16, padded, B transposed) -------
// Xh[48][1056]: row m<32 = [colsum[m]/S, surprise[m], 0...]; row 32 = [qcolsum/S, 0...]; rest 0.
__global__ void k_prep_x(const float* __restrict__ colsum, const float* __restrict__ qcolsum,
                         const float* __restrict__ surprise, _Float16* __restrict__ Xh) {
    int i = blockIdx.x * blockDim.x + threadIdx.x;
    if (i >= MPAD * KPAD) return;
    int m = i / KPAD, k = i - m * KPAD;
    float v = 0.f;
    if (k < KDIM) {
        if (m < NMEM)
            v = (k < DCOLS) ? colsum[m * DCOLS + k] * (1.f / (float)SROWS)
                            : ((k == DCOLS) ? surprise[m] : 0.f);
        else if (m == NMEM)
            v = (k < DCOLS) ? qcolsum[k] * (1.f / (float)SROWS) : 0.f;
    }
    Xh[i] = (_Float16)v;
}

// W1t[128][1056] = W1[k][n] transposed, f16, K zero-padded
__global__ void k_prep_w1t(const float* __restrict__ W1, _Float16* __restrict__ W1t) {
    int i = blockIdx.x * blockDim.x + threadIdx.x;
    if (i >= 128 * KPAD) return;
    int n = i / KPAD, k = i - n * KPAD;
    W1t[i] = (_Float16)((k < KDIM) ? W1[(size_t)k * 128 + n] : 0.f);
}

// W2t[1024][128] = W2[k][n] transposed, f16
__global__ void k_prep_w2t(const float* __restrict__ W2, _Float16* __restrict__ W2t) {
    int i = blockIdx.x * blockDim.x + threadIdx.x;
    if (i >= DCOLS * 128) return;
    int n = i >> 7, k = i & 127;
    W2t[i] = (_Float16)W2[(size_t)k * DCOLS + n];
}

// ------- meta-MLP layer 1: Xh[48,1056] @ W1t^T + b1, ReLU -> Hh[48,128] f16 via WMMA -------
// grid 24 blocks x 32 threads (1 wave per 16x16 tile). Inner loop: 4 b128 loads + wmma.
__global__ void k_meta_gemm1(const _Float16* __restrict__ Xh, const _Float16* __restrict__ W1t,
                             const float* __restrict__ b1, _Float16* __restrict__ Hh) {
    int mtile = blockIdx.x >> 3;
    int ntile = blockIdx.x & 7;
    int lane = threadIdx.x;
    int hi  = lane >> 4;
    int l15 = lane & 15;
    const _Float16* arow = Xh  + (size_t)(mtile * 16 + l15) * KPAD;
    const _Float16* brow = W1t + (size_t)(ntile * 16 + l15) * KPAD;
    v8f c = {};
    for (int kc = 0; kc < KPAD / 32; ++kc) {
        int kb = kc * 32 + hi * 8;
        v8h a0 = *reinterpret_cast<const v8h*>(arow + kb);
        v8h a1 = *reinterpret_cast<const v8h*>(arow + kb + 16);
        v8h b0 = *reinterpret_cast<const v8h*>(brow + kb);
        v8h b1v = *reinterpret_cast<const v8h*>(brow + kb + 16);
        v16h a, b;
#pragma unroll
        for (int j = 0; j < 8; ++j) {
            a[j] = a0[j]; a[j + 8] = a1[j];
            b[j] = b0[j]; b[j + 8] = b1v[j];
        }
        c = __builtin_amdgcn_wmma_f32_16x16x32_f16(false, a, false, b, (short)0, c, false, false);
    }
    int nn = ntile * 16 + l15;
#pragma unroll
    for (int r = 0; r < 8; ++r) {
        int mo = mtile * 16 + hi * 8 + r;
        Hh[mo * 128 + nn] = (_Float16)fmaxf(c[r] + b1[nn], 0.f);
    }
}

// ------- meta-MLP layer 2: Hh[48,128] @ W2t^T + b2 -> E[48,1024] f32 via WMMA -------
// grid 192 blocks x 32 threads.
__global__ void k_meta_gemm2(const _Float16* __restrict__ Hh, const _Float16* __restrict__ W2t,
                             const float* __restrict__ b2, float* __restrict__ E) {
    int mtile = blockIdx.x >> 6;
    int ntile = blockIdx.x & 63;
    int lane = threadIdx.x;
    int hi  = lane >> 4;
    int l15 = lane & 15;
    const _Float16* arow = Hh  + (size_t)(mtile * 16 + l15) * 128;
    const _Float16* brow = W2t + (size_t)(ntile * 16 + l15) * 128;
    v8f c = {};
    for (int kc = 0; kc < 4; ++kc) {
        int kb = kc * 32 + hi * 8;
        v8h a0 = *reinterpret_cast<const v8h*>(arow + kb);
        v8h a1 = *reinterpret_cast<const v8h*>(arow + kb + 16);
        v8h b0 = *reinterpret_cast<const v8h*>(brow + kb);
        v8h b1v = *reinterpret_cast<const v8h*>(brow + kb + 16);
        v16h a, b;
#pragma unroll
        for (int j = 0; j < 8; ++j) {
            a[j] = a0[j]; a[j + 8] = a1[j];
            b[j] = b0[j]; b[j + 8] = b1v[j];
        }
        c = __builtin_amdgcn_wmma_f32_16x16x32_f16(false, a, false, b, (short)0, c, false, false);
    }
    int nn = ntile * 16 + l15;
#pragma unroll
    for (int r = 0; r < 8; ++r) {
        int mo = mtile * 16 + hi * 8 + r;
        E[(size_t)mo * DCOLS + nn] = c[r] + b2[nn];
    }
}

// ------- cosine sims * priority -------
__global__ void k_sims(const float* __restrict__ E, const float* __restrict__ pm,
                       float* __restrict__ simsOut) {
    __shared__ float sd[8], sn[8], sq[8];
    int n = blockIdx.x;
    int tid = threadIdx.x;
    const float* en = E + (size_t)n * DCOLS;
    const float* eq = E + (size_t)NMEM * DCOLS;
    float dd = 0.f, nm = 0.f, nq = 0.f;
    for (int d = tid; d < DCOLS; d += blockDim.x) {
        float a = en[d], b = eq[d];
        dd += a * b; nm += a * a; nq += b * b;
    }
    for (int o = 16; o >= 1; o >>= 1) {
        dd += __shfl_xor(dd, o, 32);
        nm += __shfl_xor(nm, o, 32);
        nq += __shfl_xor(nq, o, 32);
    }
    int wid = tid >> 5, lane = tid & 31;
    if (lane == 0) { sd[wid] = dd; sn[wid] = nm; sq[wid] = nq; }
    __syncthreads();
    if (tid == 0) {
        float td = 0.f, tn = 0.f, tq = 0.f;
        for (int w = 0; w < 8; ++w) { td += sd[w]; tn += sn[w]; tq += sq[w]; }
        float s = td / (fmaxf(sqrtf(tn), 1e-8f) * fmaxf(sqrtf(tq), 1e-8f));
        simsOut[n] = s * pm[n];
    }
}

// ------- rank sort of 32 sims (descending, ties -> lower index), indices as float -------
__global__ void k_rank(const float* __restrict__ sims, float* __restrict__ idxOut) {
    int i = threadIdx.x;
    if (i < NMEM) {
        float si = sims[i];
        int rank = 0;
        for (int j = 0; j < NMEM; ++j) {
            float sj = sims[j];
            if (sj > si || (sj == si && j < i)) rank++;
        }
        idxOut[rank] = (float)i;
    }
}

extern "C" void kernel_launch(void* const* d_in, const int* in_sizes, int n_in,
                              void* d_out, int out_size, void* d_ws, size_t ws_size,
                              hipStream_t stream) {
    const float* query    = (const float*)d_in[0];
    const float* memories = (const float*)d_in[1];
    const float* surprise = (const float*)d_in[2];
    const float* pm       = (const float*)d_in[3];
    const float* W1       = (const float*)d_in[4];
    const float* b1       = (const float*)d_in[5];
    const float* W2       = (const float*)d_in[6];
    const float* b2       = (const float*)d_in[7];
    float* out = (float*)d_out;

    float* ws = (float*)d_ws;
    size_t off = 0;
    float* entropy = ws + off; off += (size_t)NMEM * SROWS;   // 65536
    float* selmask = ws + off; off += (size_t)NMEM * SROWS;   // 65536
    float* colsum  = ws + off; off += (size_t)NMEM * DCOLS;   // 32768
    float* qcolsum = ws + off; off += DCOLS;                  // contiguous after colsum
    float* agg     = ws + off; off += (size_t)SROWS * DCOLS;  // 2097152
    float* aggent  = ws + off; off += SROWS;
    float* outsel  = ws + off; off += SROWS;
    float* Ebuf    = ws + off; off += (size_t)MPAD * DCOLS;
    _Float16* Xh   = (_Float16*)(ws + off); off += (MPAD * KPAD + 1) / 2;
    _Float16* W1t  = (_Float16*)(ws + off); off += (128 * KPAD + 1) / 2;
    _Float16* W2t  = (_Float16*)(ws + off); off += (DCOLS * 128 + 1) / 2;
    _Float16* Hh   = (_Float16*)(ws + off); off += (MPAD * 128 + 1) / 2;

    // zero the atomic accumulators (colsum + qcolsum are contiguous)
    int nz = NMEM * DCOLS + DCOLS;
    k_zero<<<(nz + 255) / 256, 256, 0, stream>>>(colsum, nz);

    // single fused pass over the 256MB memories tensor (the bandwidth floor)
    k_rowstats<<<NMEM * (SROWS / 32), 256, 0, stream>>>(memories, entropy, colsum);
    k_qcolsum<<<64, 256, 0, stream>>>(query, qcolsum);

    k_topk_rows<<<NMEM, 256, 0, stream>>>(entropy, selmask, SROWS, KPATCH);
    k_aggregate<<<SROWS, 256, 0, stream>>>(query, memories, selmask, agg, aggent);
    k_topk_rows<<<1, 256, 0, stream>>>(aggent, outsel, SROWS, KPATCH);
    k_recon<<<SROWS, 256, 0, stream>>>(agg, outsel, out);

    // WMMA operand panels (f16, transposed B, zero-padded K)
    k_prep_x  <<<(MPAD * KPAD + 255) / 256, 256, 0, stream>>>(colsum, qcolsum, surprise, Xh);
    k_prep_w1t<<<(128 * KPAD + 255) / 256, 256, 0, stream>>>(W1, W1t);
    k_prep_w2t<<<(DCOLS * 128 + 255) / 256, 256, 0, stream>>>(W2, W2t);

    // meta-MLP on v_wmma_f32_16x16x32_f16 (tight load+wmma loops)
    k_meta_gemm1<<<24, 32, 0, stream>>>(Xh, W1t, b1, Hh);
    k_meta_gemm2<<<192, 32, 0, stream>>>(Hh, W2t, b2, Ebuf);

    float* simsOut = out + (size_t)SROWS * DCOLS;
    k_sims<<<NMEM, 256, 0, stream>>>(Ebuf, pm, simsOut);
    k_rank<<<1, 32, 0, stream>>>(simsOut, simsOut + NMEM);
}
